// Mamba2_9036611190978
// MI455X (gfx1250) — compile-verified, matrix operand order checked by reading, not run on hardware
//
#include <hip/hip_runtime.h>
#include <hip/hip_bf16.h>
#include <math.h>

#define DEV __device__ __forceinline__

typedef __bf16 bf16_t;
typedef __attribute__((ext_vector_type(16))) __bf16 v16bf;
typedef __attribute__((ext_vector_type(8)))  float  v8f;

// ---- problem dims ----
constexpr int D_MODEL  = 2048;
constexpr int D_STATE  = 128;
constexpr int D_CONV   = 4;
constexpr int HEADDIM  = 64;
constexpr int CHUNK    = 64;
constexpr int D_INNER  = 4096;
constexpr int NHEADS   = 64;                         // D_INNER / HEADDIM
constexpr int CONV_DIM = D_INNER + 2 * D_STATE;      // 4352
constexpr int D_IN_PROJ = 2 * D_INNER + 2 * D_STATE + NHEADS; // 8512
constexpr int BATCH    = 2;
constexpr int SEQLEN   = 2048;
constexpr int NCHUNK   = SEQLEN / CHUNK;             // 32
constexpr int ROWS     = BATCH * SEQLEN;             // 4096 tokens

// ---- helpers ----
DEV bf16_t f2bf(float f) {
  union { float f; unsigned u; } v; v.f = f;
  unsigned u = v.u;
  unsigned r = u + 0x7FFFu + ((u >> 16) & 1u);       // round-to-nearest-even
  unsigned short h = (unsigned short)(r >> 16);
  bf16_t o;
  __builtin_memcpy(&o, &h, sizeof(o));
  return o;
}

DEV float siluf(float x) { return x / (1.f + expf(-x)); }

// A-matrix / B-matrix fragment load from LDS, layout per ISA 7.12.2
// (16-bit A 16x32): lanes 0-15 row M=lane hold K [0..7],[16..23];
// lanes 16-31 row M=lane-16 hold K [8..15],[24..31].
// B fragment uses the identical pattern with rows indexed by N and K
// contiguous, so operands must be staged K-contiguous per row.
DEV v16bf frag_load(const bf16_t* buf, int stride, int row0, int koff) {
  int lane = threadIdx.x & 31;
  int row  = row0 + (lane & 15);
  int k0   = koff + ((lane >> 4) << 3);              // +0 or +8
  const bf16_t* p = buf + row * stride + k0;
  v16bf r;
#pragma unroll
  for (int j = 0; j < 8; ++j) { r[j] = p[j]; r[j + 8] = p[16 + j]; }
  return r;
}

DEV v8f wmma_bf16(v16bf a, v16bf b, v8f c) {
  return __builtin_amdgcn_wmma_f32_16x16x32_bf16(
      false, a, false, b, (short)0, c, false, false);
}

// =====================================================================
// C[M,N] = A[M,K] @ B[K,N], fp32 in/out, bf16 WMMA compute.
// 128x128 block tile, BK=64, 256 threads = 8 waves; wave w owns rows
// [16w,16w+16) x all 128 cols (8 accum tiles).
// Requires: M % 128 == 0, K % 64 == 0 (true for both call sites).
// N may be ragged (last block column handled by clamp+zero).
// =====================================================================
__global__ __launch_bounds__(256) void gemm_bf16_wmma(
    const float* __restrict__ A, const float* __restrict__ B,
    float* __restrict__ C, int M, int N, int K) {
  __shared__ bf16_t As[128 * 64];   // [m][k] k-contiguous, 16 KB
  __shared__ bf16_t Bs[128 * 64];   // [n][k] k-contiguous (B transposed), 16 KB
  const int t    = threadIdx.x;
  const int wave = t >> 5;
  const int m0   = blockIdx.y * 128;
  const int n0   = blockIdx.x * 128;

  // B-column clamp (branch-free ragged-N handling)
  const int  nb    = t & 127;            // staged column within tile
  const int  gn    = n0 + nb;
  const int  gnc   = (gn < N) ? gn : (N - 1);
  const float nmul = (gn < N) ? 1.f : 0.f;

  v8f acc[8] = {};

  for (int k0 = 0; k0 < K; k0 += 64) {
    // ---- stage A tile: thread -> (row, 32-wide k half); float4 loads ----
    {
      int r   = t >> 1;
      int kh  = (t & 1) * 32;
      const float4* src =
          (const float4*)(A + (long long)(m0 + r) * K + k0 + kh);
      bf16_t* dst = As + r * 64 + kh;
      __builtin_prefetch(src + 16, 0, 1);            // next k-tile
#pragma unroll
      for (int i = 0; i < 8; ++i) {
        float4 v = src[i];
        dst[4 * i + 0] = f2bf(v.x);
        dst[4 * i + 1] = f2bf(v.y);
        dst[4 * i + 2] = f2bf(v.z);
        dst[4 * i + 3] = f2bf(v.w);
      }
    }
    // ---- stage B tile transposed: Bs[n][k] = B[k0+k][n0+n], branch-free ----
    {
      int kh = (t >> 7) * 32;
      const float* src = B + (long long)(k0 + kh) * N + gnc;
      bf16_t* dst = Bs + nb * 64 + kh;
#pragma unroll
      for (int kk = 0; kk < 32; ++kk)
        dst[kk] = f2bf(src[(long long)kk * N] * nmul);
    }
    __syncthreads();
#pragma unroll
    for (int ks = 0; ks < 2; ++ks) {
      v16bf af = frag_load(As, 64, wave * 16, ks * 32);
#pragma unroll
      for (int nt = 0; nt < 8; ++nt) {
        v16bf bf_ = frag_load(Bs, 64, nt * 16, ks * 32);
        acc[nt] = wmma_bf16(af, bf_, acc[nt]);
      }
    }
    __syncthreads();
  }

  // epilogue: C/D layout — lane<16: N=lane, M=r; lane>=16: N=lane-16, M=8+r
  const int lane = t & 31;
  const int cn   = lane & 15;
  const int rb   = (lane >> 4) * 8;
#pragma unroll
  for (int nt = 0; nt < 8; ++nt) {
    int col = n0 + nt * 16 + cn;
#pragma unroll
    for (int r = 0; r < 8; ++r) {
      int row = m0 + wave * 16 + rb + r;
      if (col < N) C[(long long)row * N + col] = acc[nt][r];
    }
  }
}

// =====================================================================
// Depthwise causal conv (width 4) + bias + SiLU over xBC slice of zxbcdt
// =====================================================================
__global__ void conv_silu_kernel(const float* __restrict__ zxbcdt,
                                 const float* __restrict__ conv_w,
                                 const float* __restrict__ conv_b,
                                 float* __restrict__ xbc_out) {
  long long idx = (long long)blockIdx.x * blockDim.x + threadIdx.x;
  const long long total = (long long)ROWS * CONV_DIM;
  if (idx >= total) return;
  int ch        = (int)(idx % CONV_DIM);
  long long row = idx / CONV_DIM;       // b*L + l
  int l         = (int)(row % SEQLEN);
  float acc = conv_b[ch];
#pragma unroll
  for (int w = 0; w < D_CONV; ++w) {
    int ls = l - (D_CONV - 1) + w;
    float wv = (ls >= 0) ? conv_w[ch * D_CONV + w] : 0.f;
    long long lc = (ls >= 0) ? (long long)(ls - l) : 0;
    acc += zxbcdt[(row + lc) * D_IN_PROJ + D_INNER + ch] * wv;
  }
  xbc_out[idx] = siluf(acc);
}

// dt = softplus(dt_raw + dt_bias), layout [row][h]
__global__ void dt_softplus_kernel(const float* __restrict__ zxbcdt,
                                   const float* __restrict__ dt_bias,
                                   float* __restrict__ dt_out) {
  long long idx = (long long)blockIdx.x * blockDim.x + threadIdx.x;
  const long long total = (long long)ROWS * NHEADS;
  if (idx >= total) return;
  int h         = (int)(idx % NHEADS);
  long long row = idx / NHEADS;
  float v = zxbcdt[row * D_IN_PROJ + (D_INNER + CONV_DIM) + h] + dt_bias[h];
  dt_out[idx] = (v > 20.f) ? v : log1pf(expf(v));
}

// =====================================================================
// Per-(b,chunk,head): Y_diag and per-chunk state, all via WMMA.
// block = 128 threads = 4 waves; wave w owns 16-row output strip.
// =====================================================================
__global__ __launch_bounds__(128) void ssd_chunk_kernel(
    const float* __restrict__ xbc, const float* __restrict__ dt,
    const float* __restrict__ A_log, float* __restrict__ y,
    float* __restrict__ states) {
  __shared__ float  s_dt[CHUNK], s_ac[CHUNK];
  __shared__ bf16_t sC[CHUNK * D_STATE];  // C chunk  [l][n]
  __shared__ bf16_t sB[CHUNK * D_STATE];  // phase1: B [l][n]; phase2: Bdec^T [n][l]
  __shared__ bf16_t sG[CHUNK * CHUNK];    // decay-masked C·B^T, [l][s]
  __shared__ bf16_t sX[CHUNK * CHUNK];    // (x*dt)^T   [p][l]

  const int t    = threadIdx.x;
  const int wave = t >> 5;
  const int h = blockIdx.x % NHEADS;
  const int c = (blockIdx.x / NHEADS) % NCHUNK;
  const int b = blockIdx.x / (NHEADS * NCHUNK);
  const long long rowbase = (long long)b * SEQLEN + (long long)c * CHUNK;

  if (t < CHUNK) s_dt[t] = dt[(rowbase + t) * NHEADS + h];
  __syncthreads();
  if (t == 0) {                              // cumsum(A*dt) over the chunk
    float a = -expf(A_log[h]);
    float s = 0.f;
    for (int l = 0; l < CHUNK; ++l) { s += a * s_dt[l]; s_ac[l] = s; }
  }
  __syncthreads();

  // stage C,B (K = state dim, contiguous)
  for (int i = t; i < CHUNK * D_STATE; i += 128) {
    int l = i >> 7, n = i & 127;
    const float* base = xbc + (rowbase + l) * CONV_DIM;
    sC[i] = f2bf(base[D_INNER + D_STATE + n]);
    sB[i] = f2bf(base[D_INNER + n]);
  }
  __syncthreads();

  // G = C @ B^T  (64x64, K=128)
  v8f accG[4] = {};
#pragma unroll
  for (int ks = 0; ks < 4; ++ks) {
    v16bf af = frag_load(sC, D_STATE, wave * 16, ks * 32);
#pragma unroll
    for (int st = 0; st < 4; ++st) {
      v16bf bf_ = frag_load(sB, D_STATE, st * 16, ks * 32);
      accG[st] = wmma_bf16(af, bf_, accG[st]);
    }
  }
  const int lane = t & 31;
  const int cn = lane & 15, rb = (lane >> 4) * 8;
#pragma unroll
  for (int st = 0; st < 4; ++st) {
#pragma unroll
    for (int r = 0; r < 8; ++r) {
      int Mr = wave * 16 + rb + r;
      int Sc = st * 16 + cn;
      float v = (Sc <= Mr) ? accG[st][r] * expf(s_ac[Mr] - s_ac[Sc]) : 0.f;
      sG[Mr * CHUNK + Sc] = f2bf(v);
    }
  }
  __syncthreads();   // all waves done reading sB / writing sG

  // stage (x*dt)^T and decayed-B^T (reuse sB as [n][l])
  for (int i = t; i < CHUNK * CHUNK; i += 128) {
    int l = i >> 6, p = i & 63;
    float xv = xbc[(rowbase + l) * CONV_DIM + h * HEADDIM + p];
    sX[p * CHUNK + l] = f2bf(xv * s_dt[l]);
  }
  for (int i = t; i < CHUNK * D_STATE; i += 128) {
    int l = i >> 7, n = i & 127;
    float bv = xbc[(rowbase + l) * CONV_DIM + D_INNER + n];
    sB[n * CHUNK + l] = f2bf(bv * expf(s_ac[CHUNK - 1] - s_ac[l]));
  }
  __syncthreads();

  // Y_diag = G @ (x*dt)   [l][p], K = 64
  v8f accY[4] = {};
#pragma unroll
  for (int ks = 0; ks < 2; ++ks) {
    v16bf af = frag_load(sG, CHUNK, wave * 16, ks * 32);
#pragma unroll
    for (int pt = 0; pt < 4; ++pt) {
      v16bf bf_ = frag_load(sX, CHUNK, pt * 16, ks * 32);
      accY[pt] = wmma_bf16(af, bf_, accY[pt]);
    }
  }
#pragma unroll
  for (int pt = 0; pt < 4; ++pt) {
#pragma unroll
    for (int r = 0; r < 8; ++r) {
      int lrow = wave * 16 + rb + r;
      int p    = pt * 16 + cn;
      y[(rowbase + lrow) * D_INNER + h * HEADDIM + p] = accY[pt][r];
    }
  }

  // chunk state[p][n] = (x*dt)^T @ (B*decay)   K = 64
  v8f accS[8] = {};
#pragma unroll
  for (int ks = 0; ks < 2; ++ks) {
    v16bf af = frag_load(sX, CHUNK, wave * 16, ks * 32);
#pragma unroll
    for (int nt = 0; nt < 8; ++nt) {
      v16bf bf_ = frag_load(sB, CHUNK, nt * 16, ks * 32);
      accS[nt] = wmma_bf16(af, bf_, accS[nt]);
    }
  }
  const long long sbase =
      (((long long)b * NCHUNK + c) * NHEADS + h) * (HEADDIM * D_STATE);
#pragma unroll
  for (int nt = 0; nt < 8; ++nt) {
#pragma unroll
    for (int r = 0; r < 8; ++r) {
      int p = wave * 16 + rb + r;
      int n = nt * 16 + cn;
      states[sbase + p * D_STATE + n] = accS[nt][r];
    }
  }
}

// =====================================================================
// Inter-chunk scan per (b,h): replace chunk states with prefix states
// (state entering each chunk), in place.
// =====================================================================
__global__ __launch_bounds__(256) void state_scan_kernel(
    const float* __restrict__ dt, const float* __restrict__ A_log,
    float* __restrict__ states) {
  __shared__ float s_dA[NCHUNK];
  const int t = threadIdx.x;
  const int h = blockIdx.x % NHEADS;
  const int b = blockIdx.x / NHEADS;
  if (t < NCHUNK) {
    float a = -expf(A_log[h]);
    float s = 0.f;
    long long rb = (long long)b * SEQLEN + (long long)t * CHUNK;
    for (int l = 0; l < CHUNK; ++l) s += dt[(rb + l) * NHEADS + h];
    s_dA[t] = a * s;
  }
  __syncthreads();
  float run[32];
#pragma unroll
  for (int j = 0; j < 32; ++j) run[j] = 0.f;
  for (int c = 0; c < NCHUNK; ++c) {
    float ef = expf(s_dA[c]);
    long long base =
        (((long long)b * NCHUNK + c) * NHEADS + h) * (HEADDIM * D_STATE);
#pragma unroll
    for (int j = 0; j < 32; ++j) {
      long long idx = base + t + 256 * j;
      float tmp = states[idx];
      states[idx] = run[j];
      run[j] = run[j] * ef + tmp;
    }
  }
}

// =====================================================================
// Y_off = (C * exp(Acum)) @ prefix_state^T, plus skip term x*D. WMMA.
// =====================================================================
__global__ __launch_bounds__(128) void ssd_yoff_kernel(
    const float* __restrict__ xbc, const float* __restrict__ dt,
    const float* __restrict__ A_log, const float* __restrict__ D_param,
    const float* __restrict__ states, float* __restrict__ y) {
  __shared__ float  s_dt[CHUNK], s_ac[CHUNK];
  __shared__ bf16_t sCd[CHUNK * D_STATE];    // (C*outdecay) [l][n]
  __shared__ bf16_t sSt[HEADDIM * D_STATE];  // prefix state [p][n]

  const int t    = threadIdx.x;
  const int wave = t >> 5;
  const int h = blockIdx.x % NHEADS;
  const int c = (blockIdx.x / NHEADS) % NCHUNK;
  const int b = blockIdx.x / (NHEADS * NCHUNK);
  const long long rowbase = (long long)b * SEQLEN + (long long)c * CHUNK;

  if (t < CHUNK) s_dt[t] = dt[(rowbase + t) * NHEADS + h];
  __syncthreads();
  if (t == 0) {
    float a = -expf(A_log[h]);
    float s = 0.f;
    for (int l = 0; l < CHUNK; ++l) { s += a * s_dt[l]; s_ac[l] = s; }
  }
  __syncthreads();

  for (int i = t; i < CHUNK * D_STATE; i += 128) {
    int l = i >> 7, n = i & 127;
    float cv = xbc[(rowbase + l) * CONV_DIM + D_INNER + D_STATE + n];
    sCd[i] = f2bf(cv * expf(s_ac[l]));
  }
  const long long sbase =
      (((long long)b * NCHUNK + c) * NHEADS + h) * (HEADDIM * D_STATE);
  for (int i = t; i < HEADDIM * D_STATE; i += 128) sSt[i] = f2bf(states[sbase + i]);
  __syncthreads();

  v8f accO[4] = {};
#pragma unroll
  for (int ks = 0; ks < 4; ++ks) {
    v16bf af = frag_load(sCd, D_STATE, wave * 16, ks * 32);
#pragma unroll
    for (int pt = 0; pt < 4; ++pt) {
      v16bf bf_ = frag_load(sSt, D_STATE, pt * 16, ks * 32);
      accO[pt] = wmma_bf16(af, bf_, accO[pt]);
    }
  }
  const int lane = t & 31;
  const int cn = lane & 15, rb = (lane >> 4) * 8;
  const float Dh = D_param[h];
#pragma unroll
  for (int pt = 0; pt < 4; ++pt) {
#pragma unroll
    for (int r = 0; r < 8; ++r) {
      int lrow = wave * 16 + rb + r;
      int p    = pt * 16 + cn;
      long long yi = (rowbase + lrow) * D_INNER + h * HEADDIM + p;
      float xv = xbc[(rowbase + lrow) * CONV_DIM + h * HEADDIM + p];
      y[yi] = y[yi] + accO[pt][r] + xv * Dh;
    }
  }
}

// =====================================================================
// Gated RMSNorm: yf = y * silu(z); out = yf * rsqrt(mean(yf^2)+eps) * w
// =====================================================================
__global__ __launch_bounds__(256) void gated_norm_kernel(
    const float* __restrict__ y, const float* __restrict__ zxbcdt,
    const float* __restrict__ norm_w, float* __restrict__ out) {
  __shared__ float red[256];
  const long long row = blockIdx.x;
  const int t = threadIdx.x;
  const float* yrow = y + row * D_INNER;
  const float* zrow = zxbcdt + row * D_IN_PROJ;   // z = first D_INNER cols
  float vals[16];
  float ss = 0.f;
#pragma unroll
  for (int j = 0; j < 16; ++j) {
    int ch = t + 256 * j;
    float yf = yrow[ch] * siluf(zrow[ch]);
    vals[j] = yf;
    ss += yf * yf;
  }
  red[t] = ss;
  __syncthreads();
  for (int off = 128; off > 0; off >>= 1) {
    if (t < off) red[t] += red[t + off];
    __syncthreads();
  }
  float rrms = rsqrtf(red[0] / (float)D_INNER + 1e-5f);
#pragma unroll
  for (int j = 0; j < 16; ++j) {
    int ch = t + 256 * j;
    out[row * D_INNER + ch] = vals[j] * rrms * norm_w[ch];
  }
}

// =====================================================================
extern "C" void kernel_launch(void* const* d_in, const int* in_sizes, int n_in,
                              void* d_out, int out_size, void* d_ws,
                              size_t ws_size, hipStream_t stream) {
  (void)in_sizes; (void)n_in; (void)out_size; (void)ws_size;
  const float* u       = (const float*)d_in[0];
  const float* W_in    = (const float*)d_in[1];
  const float* conv_w  = (const float*)d_in[2];
  const float* conv_b  = (const float*)d_in[3];
  const float* dt_bias = (const float*)d_in[4];
  const float* A_log   = (const float*)d_in[5];
  const float* D_param = (const float*)d_in[6];
  const float* norm_w  = (const float*)d_in[7];
  const float* W_out   = (const float*)d_in[8];
  float* out = (float*)d_out;

  float* ws = (float*)d_ws;
  long long off = 0;
  float* zx     = ws + off; off += (long long)ROWS * D_IN_PROJ;   // 139.5 MB
  float* xbc    = ws + off; off += (long long)ROWS * CONV_DIM;    //  71.3 MB
  float* dtbuf  = ws + off; off += (long long)ROWS * NHEADS;      //   1.0 MB
  float* states = ws + off;
  off += (long long)BATCH * NCHUNK * NHEADS * HEADDIM * D_STATE;  // 134.2 MB
  float* ybuf   = ws + off; off += (long long)ROWS * D_INNER;     //  67.1 MB
  float* normed = ws + off; off += (long long)ROWS * D_INNER;     //  67.1 MB

  // 1) in_proj: [4096,2048] @ [2048,8512]
  gemm_bf16_wmma<<<dim3((D_IN_PROJ + 127) / 128, ROWS / 128), 256, 0, stream>>>(
      u, W_in, zx, ROWS, D_IN_PROJ, D_MODEL);

  // 2) conv + silu
  {
    long long total = (long long)ROWS * CONV_DIM;
    conv_silu_kernel<<<(unsigned)((total + 255) / 256), 256, 0, stream>>>(
        zx, conv_w, conv_b, xbc);
  }
  // 3) dt softplus
  {
    long long total = (long long)ROWS * NHEADS;
    dt_softplus_kernel<<<(unsigned)((total + 255) / 256), 256, 0, stream>>>(
        zx, dt_bias, dtbuf);
  }
  // 4) per-chunk SSD (Y_diag + chunk states)
  ssd_chunk_kernel<<<BATCH * NCHUNK * NHEADS, 128, 0, stream>>>(
      xbc, dtbuf, A_log, ybuf, states);
  // 5) inter-chunk state scan (in place -> prefix states)
  state_scan_kernel<<<BATCH * NHEADS, 256, 0, stream>>>(dtbuf, A_log, states);
  // 6) Y_off + skip connection
  ssd_yoff_kernel<<<BATCH * NCHUNK * NHEADS, 128, 0, stream>>>(
      xbc, dtbuf, A_log, D_param, states, ybuf);
  // 7) gated RMSNorm
  gated_norm_kernel<<<ROWS, 256, 0, stream>>>(ybuf, zx, norm_w, normed);
  // 8) out_proj: [4096,4096] @ [4096,2048]
  gemm_bf16_wmma<<<dim3(D_MODEL / 128, ROWS / 128), 256, 0, stream>>>(
      normed, W_out, out, ROWS, D_MODEL, D_INNER);
}
